// FSRAVMKModel_86887188398946
// MI455X (gfx1250) — matrix operand order repaired and checked by reading, not compiled, not run on hardware
//
#include <hip/hip_runtime.h>
#include <math.h>

typedef __bf16 bf16_t;
typedef __attribute__((ext_vector_type(16))) __bf16 v16bf;
typedef __attribute__((ext_vector_type(8)))  float  v8f;

// ---------------- device helpers ----------------

__device__ inline float wave_sum(float v) {
#pragma unroll
  for (int off = 16; off; off >>= 1) v += __shfl_xor(v, off, 32);
  return v;
}
__device__ inline float wave_max(float v) {
#pragma unroll
  for (int off = 16; off; off >>= 1) v = fmaxf(v, __shfl_xor(v, off, 32));
  return v;
}

union FragU { uint4 u[2]; v16bf v; };

// A fragment: 16x32 bf16, lane holds row M=lane%16; element e -> K = (lane&16) + e
// -> 16 contiguous halves (pointer pre-offset by lane&16).
__device__ inline v16bf frag_a(const bf16_t* p) {
  FragU f;
  f.u[0] = *reinterpret_cast<const uint4*>(p);
  f.u[1] = *reinterpret_cast<const uint4*>(p + 8);
  return f.v;
}
// B fragment: 32x16 bf16, lane holds col N=lane%16; element e -> K = (lane&16) + e
// -> 16 contiguous halves (pointer pre-offset by lane&16).
__device__ inline v16bf frag_b(const bf16_t* p) {
  FragU f;
  f.u[0] = *reinterpret_cast<const uint4*>(p);
  f.u[1] = *reinterpret_cast<const uint4*>(p + 8);
  return f.v;
}
__device__ inline v8f wmma_bf16(v16bf a, v16bf b, v8f c) {
  return __builtin_amdgcn_wmma_f32_16x16x32_bf16(false, a, false, b, (short)0, c,
                                                 false, false);
}

// Async global->LDS copy of 16 bytes per lane (gfx1250, tracked by ASYNCcnt).
__device__ inline void async_b128(unsigned lds_off, unsigned long long gaddr) {
  asm volatile("global_load_async_to_lds_b128 %0, %1, off"
               :: "v"(lds_off), "v"(gaddr)
               : "memory");
}
__device__ inline void wait_async0() {
  asm volatile("s_wait_asynccnt 0" ::: "memory");
}
// Low 32 bits of a generic pointer into __shared__ are the LDS byte offset.
__device__ inline unsigned lds_off_of(const void* p) {
  return (unsigned)(unsigned long long)(uintptr_t)p;
}

__device__ inline float gelu_f(float x) {
  return 0.5f * x * (1.0f + erff(x * 0.70710678118654752440f));
}

// normalized Gaussian RBF bases on grid linspace(-1,1,6); INV2SIG2 = 12.5
__device__ inline void kan_bases(float x, float* bb) {
  float s = 0.f;
#pragma unroll
  for (int g = 0; g < 6; ++g) {
    float d = x - (-1.0f + 0.4f * (float)g);
    bb[g] = expf(-d * d * 12.5f);
    s += bb[g];
  }
  float inv = 1.0f / (s + 1e-8f);
#pragma unroll
  for (int g = 0; g < 6; ++g) bb[g] *= inv;
}

// ---------------- kernels ----------------

// Pack KAN weights (out,in)+(out,in,6) -> bf16 (out, in*7) row-major.
__global__ void k_pack_w(const float* __restrict__ bw, const float* __restrict__ sw,
                         bf16_t* __restrict__ Wp, long long total) {
  long long idx = (long long)blockIdx.x * blockDim.x + threadIdx.x;
  if (idx >= total) return;
  bf16_t* w = Wp + idx * 7;
  w[0] = (bf16_t)bw[idx];
  const float* s = sw + idx * 6;
#pragma unroll
  for (int j = 0; j < 6; ++j) w[1 + j] = (bf16_t)s[j];
}

// (B,C,H,W) f32 -> (B, N, C) f32
__global__ void k_nchw2ntc(const float* __restrict__ in, float* __restrict__ out) {
  long long idx = (long long)blockIdx.x * blockDim.x + threadIdx.x;
  if (idx >= 8LL * 576 * 256) return;
  int c = (int)(idx & 255);
  long long t = idx >> 8;
  int n = (int)(t % 576), b = (int)(t / 576);
  out[idx] = in[((size_t)b * 256 + c) * 576 + n];
}

// X (T, in) f32 -> E (T, in*7) bf16
__global__ void k_expand(const float* __restrict__ X, bf16_t* __restrict__ E,
                         long long total) {
  long long idx = (long long)blockIdx.x * blockDim.x + threadIdx.x;
  if (idx >= total) return;
  float x = X[idx];
  bf16_t* e = E + idx * 7;
  e[0] = (bf16_t)x;
  float bb[6];
  kan_bases(x, bb);
#pragma unroll
  for (int g = 0; g < 6; ++g) e[1 + g] = (bf16_t)bb[g];
}

// ---- big GEMM: async global->LDS double-buffered, 128x64 macro-tile ----
#define BM 128
#define BN 64
#define LDA 40  // padded LDS row stride in halves (80B: 16B-aligned, spreads banks)

__global__ __launch_bounds__(256) void k_gemm_lds(const bf16_t* __restrict__ A,
                                                  const bf16_t* __restrict__ W,
                                                  float* __restrict__ D, int Mb,
                                                  int K, int N) {
  __shared__ bf16_t As[2][BM * LDA];
  __shared__ bf16_t Bs[2][BN * LDA];
  const int tid = threadIdx.x;
  const int bm = blockIdx.x % Mb, bn = blockIdx.x / Mb;
  const int mBase = bm * BM, nBase = bn * BN;

  // staging assignment: A has 512 16B-chunks (2 per thread), B has 256 (1 per thread)
  const int ra0 = tid >> 2, ch = tid & 3;
  const int ra1 = (tid + 256) >> 2;  // (tid+256)&3 == ch
  const int rb = tid >> 2;
  unsigned long long gA0 = (unsigned long long)(uintptr_t)(A + (size_t)(mBase + ra0) * K + ch * 8);
  unsigned long long gA1 = (unsigned long long)(uintptr_t)(A + (size_t)(mBase + ra1) * K + ch * 8);
  unsigned long long gB  = (unsigned long long)(uintptr_t)(W + (size_t)(nBase + rb) * K + ch * 8);
  unsigned lA0[2], lA1[2], lB[2];
#pragma unroll
  for (int b = 0; b < 2; ++b) {
    lA0[b] = lds_off_of(&As[b][ra0 * LDA + ch * 8]);
    lA1[b] = lds_off_of(&As[b][ra1 * LDA + ch * 8]);
    lB[b] = lds_off_of(&Bs[b][rb * LDA + ch * 8]);
  }

  const int wave = tid >> 5, lane = tid & 31;
  const int wm = wave & 3, wn = wave >> 1 >> 1;  // 4 x 2 wave grid
  const int lo8 = (lane & 16) >> 1, hi16 = lane & 16;

  v8f acc00 = {}, acc01 = {}, acc10 = {}, acc11 = {};

  // prologue: stage k-block 0 into buffer 0
  async_b128(lA0[0], gA0);
  async_b128(lA1[0], gA1);
  async_b128(lB[0], gB);
  gA0 += 64; gA1 += 64; gB += 64;
  wait_async0();
  __syncthreads();

  const int nk = K / 32;
  for (int ks = 0; ks < nk; ++ks) {
    const int cur = ks & 1;
    if (ks + 1 < nk) {  // stage next k-block into other buffer
      const int nxt = cur ^ 1;
      async_b128(lA0[nxt], gA0);
      async_b128(lA1[nxt], gA1);
      async_b128(lB[nxt], gB);
      gA0 += 64; gA1 += 64; gB += 64;
    }
    // fragments from LDS (ds_load_b128 pairs)
    const bf16_t* a0p = &As[cur][(wm * 32 + (lane & 15)) * LDA + hi16];
    const bf16_t* b0p = &Bs[cur][(wn * 32 + (lane & 15)) * LDA + hi16];
    v16bf a0 = frag_a(a0p);
    v16bf a1 = frag_a(a0p + 16 * LDA);
    v16bf b0 = frag_b(b0p);
    v16bf b1 = frag_b(b0p + 16 * LDA);
    acc00 = wmma_bf16(a0, b0, acc00);
    acc01 = wmma_bf16(a0, b1, acc01);
    acc10 = wmma_bf16(a1, b0, acc10);
    acc11 = wmma_bf16(a1, b1, acc11);
    wait_async0();
    __syncthreads();
  }

  const int m0 = mBase + wm * 32 + lo8;
  const int c0 = nBase + wn * 32 + (lane & 15);
#pragma unroll
  for (int r = 0; r < 8; ++r) {
    float* d0 = D + (size_t)(m0 + r) * N + c0;
    float* d1 = D + (size_t)(m0 + 16 + r) * N + c0;
    d0[0] = acc00[r];
    d0[16] = acc01[r];
    d1[0] = acc10[r];
    d1[16] = acc11[r];
  }
}

// small GEMM (head, M=16): direct-global 16x64 per wave
__global__ __launch_bounds__(256) void k_gemm(const bf16_t* __restrict__ A,
                                              const bf16_t* __restrict__ W,
                                              float* __restrict__ D, int M16,
                                              int N64, int K, int N) {
  int wid = blockIdx.x * 8 + (threadIdx.x >> 5);
  if (wid >= M16 * N64) return;
  int lane = threadIdx.x & 31;
  int mt = wid % M16, nt = wid / M16;
  int lo8 = (lane & 16) >> 1;
  int hi16 = lane & 16;
  const bf16_t* aRow = A + (size_t)(mt * 16 + (lane & 15)) * K + hi16;
  const bf16_t* bCol = W + (size_t)(nt * 64 + (lane & 15)) * K + hi16;
  const size_t K16 = (size_t)16 * K;
  v8f acc0 = {}, acc1 = {}, acc2 = {}, acc3 = {};
  for (int kb = 0; kb < K; kb += 32) {
    v16bf a = frag_a(aRow + kb);
    acc0 = wmma_bf16(a, frag_b(bCol + kb), acc0);
    acc1 = wmma_bf16(a, frag_b(bCol + K16 + kb), acc1);
    acc2 = wmma_bf16(a, frag_b(bCol + 2 * K16 + kb), acc2);
    acc3 = wmma_bf16(a, frag_b(bCol + 3 * K16 + kb), acc3);
  }
  int m0 = mt * 16 + lo8;
  int c0 = nt * 64 + (lane & 15);
#pragma unroll
  for (int r = 0; r < 8; ++r) {
    float* d = D + (size_t)(m0 + r) * N + c0;
    d[0] = acc0[r];
    d[16] = acc1[r];
    d[32] = acc2[r];
    d[48] = acc3[r];
  }
}

// (T,256) f32 -> bf16 heads: mode 0 -> (b,h,n,d); mode 1 -> (b,h,d,n) (V^T)
__global__ void k_split_heads(const float* __restrict__ Q, bf16_t* __restrict__ out,
                              int transposed) {
  long long idx = (long long)blockIdx.x * blockDim.x + threadIdx.x;
  if (idx >= 4608LL * 256) return;
  int c = (int)(idx & 255);
  long long t = idx >> 8;
  int n = (int)(t % 576), b = (int)(t / 576);
  int h = c >> 5, d = c & 31;
  size_t o = transposed ? (((size_t)(b * 8 + h) * 32 + d) * 576 + n)
                        : (((size_t)(b * 8 + h) * 576 + n) * 32 + d);
  out[o] = (bf16_t)Q[idx];
}

// relative-position bias: bias[h][n][m]
__global__ void k_rel_bias(const float* __restrict__ rb, const float* __restrict__ rs,
                           float* __restrict__ bias) {
  long long idx = (long long)blockIdx.x * blockDim.x + threadIdx.x;
  if (idx >= 576LL * 576) return;
  int m = (int)(idx % 576), n = (int)(idx / 576);
  float x0 = (float)((n / 24) - (m / 24)) * (1.0f / 23.0f);
  float x1 = (float)((n % 24) - (m % 24)) * (1.0f / 23.0f);
  float b0[6], b1[6];
  kan_bases(x0, b0);
  kan_bases(x1, b1);
#pragma unroll
  for (int h = 0; h < 8; ++h) {
    float acc = x0 * rb[h * 2 + 0] + x1 * rb[h * 2 + 1];
    const float* sh = rs + h * 12;
#pragma unroll
    for (int g = 0; g < 6; ++g) acc += b0[g] * sh[g] + b1[g] * sh[6 + g];
    bias[((size_t)h * 576 + n) * 576 + m] = acc;
  }
}

// S[b,h,n,m] = scale * q.k + bias[h,n,m]; one wave = one 16x16 tile (K=32, 1 WMMA)
__global__ __launch_bounds__(256) void k_scores(const bf16_t* __restrict__ qh,
                                                const bf16_t* __restrict__ kh,
                                                const float* __restrict__ bias,
                                                float* __restrict__ S, float scale) {
  int wid = blockIdx.x * 8 + (threadIdx.x >> 5);
  if (wid >= 8 * 8 * 36 * 36) return;
  int lane = threadIdx.x & 31;
  int nt = wid % 36;
  int mt = (wid / 36) % 36;
  int h = (wid / 1296) % 8;
  int b = wid / 10368;
  size_t bh = (size_t)(b * 8 + h);
  const bf16_t* qa = qh + (bh * 576 + mt * 16 + (lane & 15)) * 32 + (lane & 16);
  const bf16_t* kb = kh + (bh * 576 + nt * 16 + (lane & 15)) * 32 + (lane & 16);
  v8f acc = {};
  acc = wmma_bf16(frag_a(qa), frag_b(kb), acc);
  int m0 = mt * 16 + ((lane & 16) >> 1);
  int n = nt * 16 + (lane & 15);
  const float* bp = bias + (size_t)h * 576 * 576;
  float* sp = S + bh * 576 * 576;
#pragma unroll
  for (int r = 0; r < 8; ++r) {
    size_t o = (size_t)(m0 + r) * 576 + n;
    sp[o] = acc[r] * scale + bp[o];
  }
}

// row softmax over 576, f32 -> bf16 probs; one wave per row
__global__ __launch_bounds__(256) void k_softmax(const float* __restrict__ S,
                                                 bf16_t* __restrict__ P) {
  int row = blockIdx.x * 8 + (threadIdx.x >> 5);
  if (row >= 36864) return;
  int lane = threadIdx.x & 31;
  const float* s = S + (size_t)row * 576;
  bf16_t* p = P + (size_t)row * 576;
  float v[18], mx = -3.4e38f;
#pragma unroll
  for (int j = 0; j < 18; ++j) {
    v[j] = s[lane + j * 32];
    mx = fmaxf(mx, v[j]);
  }
  mx = wave_max(mx);
  float sum = 0.f;
#pragma unroll
  for (int j = 0; j < 18; ++j) {
    v[j] = expf(v[j] - mx);
    sum += v[j];
  }
  sum = wave_sum(sum);
  float inv = 1.0f / sum;
#pragma unroll
  for (int j = 0; j < 18; ++j) p[lane + j * 32] = (bf16_t)(v[j] * inv);
}

// O[b,n,h*32+d] = sum_m P[b,h,n,m] * V[b,h,m,d]; one wave = 16 rows x 32 cols
__global__ __launch_bounds__(256) void k_attn_out(const bf16_t* __restrict__ P,
                                                  const bf16_t* __restrict__ vT,
                                                  float* __restrict__ O) {
  int wid = blockIdx.x * 8 + (threadIdx.x >> 5);
  if (wid >= 8 * 8 * 36) return;
  int lane = threadIdx.x & 31;
  int mt = wid % 36;
  int h = (wid / 36) % 8;
  int b = wid / 288;
  size_t bh = (size_t)(b * 8 + h);
  const bf16_t* aRow = P + (bh * 576 + mt * 16 + (lane & 15)) * 576 + (lane & 16);
  const bf16_t* b0 = vT + (bh * 32 + (lane & 15)) * 576 + (lane & 16);
  const bf16_t* b1 = b0 + (size_t)16 * 576;
  v8f acc0 = {}, acc1 = {};
  for (int kb = 0; kb < 576; kb += 32) {
    v16bf a = frag_a(aRow + kb);
    acc0 = wmma_bf16(a, frag_b(b0 + kb), acc0);
    acc1 = wmma_bf16(a, frag_b(b1 + kb), acc1);
  }
  int m0 = mt * 16 + ((lane & 16) >> 1);
  int d = lane & 15;
#pragma unroll
  for (int r = 0; r < 8; ++r) {
    float* dst = O + ((size_t)b * 576 + m0 + r) * 256 + h * 32;
    dst[d] = acc0[r];
    dst[d + 16] = acc1[r];
  }
}

__global__ void k_gelu(float* __restrict__ X, long long n) {
  long long idx = (long long)blockIdx.x * blockDim.x + threadIdx.x;
  if (idx < n) X[idx] = gelu_f(X[idx]);
}

// out = se + LayerNorm(h) * g + b ; one wave per 256-wide row
__global__ __launch_bounds__(256) void k_add_ln(const float* __restrict__ se,
                                                const float* __restrict__ hin,
                                                const float* __restrict__ g,
                                                const float* __restrict__ bb,
                                                float* __restrict__ out, int rows) {
  int row = blockIdx.x * 8 + (threadIdx.x >> 5);
  if (row >= rows) return;
  int lane = threadIdx.x & 31;
  const float* x = hin + (size_t)row * 256;
  float v[8], s = 0.f;
#pragma unroll
  for (int j = 0; j < 8; ++j) {
    v[j] = x[lane + j * 32];
    s += v[j];
  }
  s = wave_sum(s);
  float m = s * (1.0f / 256.0f), q = 0.f;
#pragma unroll
  for (int j = 0; j < 8; ++j) {
    float d = v[j] - m;
    q += d * d;
  }
  q = wave_sum(q);
  float inv = rsqrtf(q * (1.0f / 256.0f) + 1e-5f);
#pragma unroll
  for (int j = 0; j < 8; ++j) {
    int c = lane + j * 32;
    out[(size_t)row * 256 + c] =
        se[(size_t)row * 256 + c] + (v[j] - m) * inv * g[c] + bb[c];
  }
}

// sg[b,c] = mean over n of X[b,n,c]
__global__ void k_mean_n(const float* __restrict__ X, float* __restrict__ sg) {
  int idx = blockIdx.x * blockDim.x + threadIdx.x;
  if (idx >= 8 * 256) return;
  int c = idx & 255, b = idx >> 8;
  float s = 0.f;
  for (int n = 0; n < 576; ++n) s += X[((size_t)b * 576 + n) * 256 + c];
  sg[idx] = s * (1.0f / 576.0f);
}

__global__ void k_make_si(const float* __restrict__ sg, const float* __restrict__ ug,
                          float* __restrict__ si) {
  int idx = blockIdx.x * blockDim.x + threadIdx.x;
  if (idx >= 16 * 512) return;
  int c = idx & 511, r = idx >> 9;
  si[idx] = (r < 8) ? ((c < 256) ? sg[r * 256 + c] : ug[r * 256 + c - 256]) : 0.f;
}

__global__ void k_make_fi(const float* __restrict__ sg, const float* __restrict__ ug,
                          const float* __restrict__ w, float* __restrict__ fi) {
  int idx = blockIdx.x * blockDim.x + threadIdx.x;
  if (idx >= 16 * 512) return;
  int c = idx & 511, r = idx >> 9;
  float v = 0.f;
  if (r < 8) {
    float wv = w[r];
    v = (c < 256) ? sg[r * 256 + c] * wv : ug[r * 256 + c - 256] * (1.0f - wv);
  }
  fi[idx] = v;
}

// w[b] = sigmoid(kan(h2[b,:64], sim_w3)); one wave per batch row
__global__ void k_simw3(const float* __restrict__ h2, const float* __restrict__ bw,
                        const float* __restrict__ sw, float* __restrict__ wout) {
  int b = blockIdx.x, lane = threadIdx.x;
  float acc = 0.f;
  for (int i = lane; i < 64; i += 32) {
    float x = h2[b * 64 + i];
    float bb[6];
    kan_bases(x, bb);
    float a = x * bw[i];
#pragma unroll
    for (int g = 0; g < 6; ++g) a += bb[g] * sw[i * 6 + g];
    acc += a;
  }
  acc = wave_sum(acc);
  if (lane == 0) wout[b] = 1.0f / (1.0f + expf(-acc));
}

// out[b,:] = gelu(LN(fusO[b,:]) * g + bb), b<8 ; one wave per row
__global__ __launch_bounds__(256) void k_final(const float* __restrict__ fusO,
                                               const float* __restrict__ g,
                                               const float* __restrict__ bb,
                                               float* __restrict__ out) {
  int row = threadIdx.x >> 5;
  int lane = threadIdx.x & 31;
  const float* x = fusO + (size_t)row * 256;
  float v[8], s = 0.f;
#pragma unroll
  for (int j = 0; j < 8; ++j) {
    v[j] = x[lane + j * 32];
    s += v[j];
  }
  s = wave_sum(s);
  float m = s * (1.0f / 256.0f), q = 0.f;
#pragma unroll
  for (int j = 0; j < 8; ++j) {
    float d = v[j] - m;
    q += d * d;
  }
  q = wave_sum(q);
  float inv = rsqrtf(q * (1.0f / 256.0f) + 1e-5f);
#pragma unroll
  for (int j = 0; j < 8; ++j) {
    int c = lane + j * 32;
    out[row * 256 + c] = gelu_f((v[j] - m) * inv * g[c] + bb[c]);
  }
}

// ---------------- host ----------------

extern "C" void kernel_launch(void* const* d_in, const int* in_sizes, int n_in,
                              void* d_out, int out_size, void* d_ws, size_t ws_size,
                              hipStream_t stream) {
  (void)in_sizes;
  (void)n_in;
  (void)out_size;
  const float* sat_feat = (const float*)d_in[0];
  const float* uav_feat = (const float*)d_in[1];
  const float* qkvp_base[2] = {(const float*)d_in[2], (const float*)d_in[6]};
  const float* qkvp_spl[2] = {(const float*)d_in[3], (const float*)d_in[7]};
  const float* rp_base[2] = {(const float*)d_in[4], (const float*)d_in[8]};
  const float* rp_spl[2] = {(const float*)d_in[5], (const float*)d_in[9]};
  const float* r1_base = (const float*)d_in[10];
  const float* r1_spl = (const float*)d_in[11];
  const float* r2_base = (const float*)d_in[12];
  const float* r2_spl = (const float*)d_in[13];
  const float* rln_g = (const float*)d_in[14];
  const float* rln_b = (const float*)d_in[15];
  const float* s1_base = (const float*)d_in[16];
  const float* s1_spl = (const float*)d_in[17];
  const float* s2_base = (const float*)d_in[18];
  const float* s2_spl = (const float*)d_in[19];
  const float* s3_base = (const float*)d_in[20];
  const float* s3_spl = (const float*)d_in[21];
  const float* fw_base = (const float*)d_in[22];
  const float* fw_spl = (const float*)d_in[23];
  const float* fln_g = (const float*)d_in[24];
  const float* fln_b = (const float*)d_in[25];

  const long long T = 4608;  // B*N
  char* p = (char*)d_ws;
  auto carve = [&](size_t bytes) {
    char* r = p;
    p += (bytes + 255) & ~(size_t)255;
    return r;
  };
  float* satX = (float*)carve(T * 256 * 4);
  float* uavX = (float*)carve(T * 256 * 4);
  bf16_t* Ebuf = (bf16_t*)carve(T * 7168 * 2);
  float* H1 = (float*)carve(T * 1024 * 4);
  float* QKVf = (float*)carve(T * 256 * 4);  // reused as attention output O
  float* seB = (float*)carve(T * 256 * 4);
  float* h2B = (float*)carve(T * 256 * 4);
  bf16_t* qh = (bf16_t*)carve(T * 256 * 2);
  bf16_t* kh = (bf16_t*)carve(T * 256 * 2);
  bf16_t* vT = (bf16_t*)carve(T * 256 * 2);
  float* Sb = (float*)carve(64LL * 576 * 576 * 4);
  bf16_t* Pb = (bf16_t*)carve(64LL * 576 * 576 * 2);
  float* biasB = (float*)carve(8LL * 576 * 576 * 4);
  bf16_t* Wqkvp = (bf16_t*)carve(24LL * 256 * 1792 * 2);
  bf16_t* Wr1 = (bf16_t*)carve(3LL * 1024 * 1792 * 2);
  bf16_t* Wr2 = (bf16_t*)carve(3LL * 256 * 7168 * 2);
  bf16_t* Ws1 = (bf16_t*)carve(1024LL * 3584 * 2);
  bf16_t* Ws2 = (bf16_t*)carve(64LL * 7168 * 2);
  bf16_t* Wfu = (bf16_t*)carve(256LL * 3584 * 2);
  float* sg = (float*)carve(8 * 256 * 4);
  float* ug = (float*)carve(8 * 256 * 4);
  float* siB = (float*)carve(16 * 512 * 4);
  bf16_t* siE = (bf16_t*)carve(16LL * 3584 * 2);
  float* h1h = (float*)carve(16 * 1024 * 4);
  bf16_t* E2h = (bf16_t*)carve(16LL * 7168 * 2);
  float* h2h = (float*)carve(16 * 64 * 4);
  float* wB = (float*)carve(8 * 4);
  float* fiB = (float*)carve(16 * 512 * 4);
  bf16_t* fiE = (bf16_t*)carve(16LL * 3584 * 2);
  float* fusO = (float*)carve(16 * 256 * 4);
  if ((size_t)(p - (char*)d_ws) > ws_size) return;  // workspace too small

  auto NB = [](long long n) { return (unsigned)((n + 255) / 256); };
  auto gemm = [&](const bf16_t* A, const bf16_t* Wt, float* D, int M, int K, int N) {
    if ((M % 128) == 0 && (N % 64) == 0) {
      int Mb = M / 128, Nb = N / 64;
      k_gemm_lds<<<Mb * Nb, 256, 0, stream>>>(A, Wt, D, Mb, K, N);
    } else {
      int waves = (M / 16) * (N / 64);
      k_gemm<<<(waves + 7) / 8, 256, 0, stream>>>(A, Wt, D, M / 16, N / 64, K, N);
    }
  };

  // ---- pack all KAN weights to bf16 (out, in*7) ----
  for (int s = 0; s < 2; ++s)
    for (int i = 0; i < 3; ++i)
      for (int sl = 0; sl < 4; ++sl) {
        long long off = ((long long)i * 4 + sl) * 256 * 256;
        k_pack_w<<<NB(256LL * 256), 256, 0, stream>>>(
            qkvp_base[s] + off, qkvp_spl[s] + off * 6,
            Wqkvp + (((long long)s * 3 + i) * 4 + sl) * (256LL * 1792), 256LL * 256);
      }
  for (int i = 0; i < 3; ++i) {
    k_pack_w<<<NB(1024LL * 256), 256, 0, stream>>>(
        r1_base + (long long)i * 1024 * 256, r1_spl + (long long)i * 1024 * 256 * 6,
        Wr1 + (long long)i * 1024 * 1792, 1024LL * 256);
    k_pack_w<<<NB(256LL * 1024), 256, 0, stream>>>(
        r2_base + (long long)i * 256 * 1024, r2_spl + (long long)i * 256 * 1024 * 6,
        Wr2 + (long long)i * 256 * 7168, 256LL * 1024);
  }
  k_pack_w<<<NB(1024LL * 512), 256, 0, stream>>>(s1_base, s1_spl, Ws1, 1024LL * 512);
  k_pack_w<<<NB(64LL * 1024), 256, 0, stream>>>(s2_base, s2_spl, Ws2, 64LL * 1024);
  k_pack_w<<<NB(256LL * 512), 256, 0, stream>>>(fw_base, fw_spl, Wfu, 256LL * 512);

  // ---- inputs to (B,N,C) ----
  k_nchw2ntc<<<NB(T * 256), 256, 0, stream>>>(sat_feat, satX);
  k_nchw2ntc<<<NB(T * 256), 256, 0, stream>>>(uav_feat, uavX);

  const float scale = 0.17677669529663687f;  // 32^-0.5
  for (int i = 0; i < 3; ++i) {
    for (int s = 0; s < 2; ++s) {
      float* X = s == 0 ? satX : uavX;
      const bf16_t* Wq = Wqkvp + ((long long)s * 3 + i) * 4 * (256LL * 1792);
      // expand X once, project q/k/v
      k_expand<<<NB(T * 256), 256, 0, stream>>>(X, Ebuf, T * 256);
      for (int sl = 0; sl < 3; ++sl) {
        gemm(Ebuf, Wq + (long long)sl * 256 * 1792, QKVf, 4608, 1792, 256);
        k_split_heads<<<NB(T * 256), 256, 0, stream>>>(
            QKVf, sl == 0 ? qh : (sl == 1 ? kh : vT), sl == 2 ? 1 : 0);
      }
      k_rel_bias<<<NB(576LL * 576), 256, 0, stream>>>(rp_base[s] + i * 16,
                                                      rp_spl[s] + i * 96, biasB);
      k_scores<<<10368, 256, 0, stream>>>(qh, kh, biasB, Sb, scale);
      k_softmax<<<4608, 256, 0, stream>>>(Sb, Pb);
      k_attn_out<<<288, 256, 0, stream>>>(Pb, vT, QKVf);
      // output projection
      k_expand<<<NB(T * 256), 256, 0, stream>>>(QKVf, Ebuf, T * 256);
      gemm(Ebuf, Wq + 3LL * 256 * 1792, seB, 4608, 1792, 256);
      // refine
      k_expand<<<NB(T * 256), 256, 0, stream>>>(seB, Ebuf, T * 256);
      gemm(Ebuf, Wr1 + (long long)i * 1024 * 1792, H1, 4608, 1792, 1024);
      k_gelu<<<NB(T * 1024), 256, 0, stream>>>(H1, T * 1024);
      k_expand<<<NB(T * 1024), 256, 0, stream>>>(H1, Ebuf, T * 1024);
      gemm(Ebuf, Wr2 + (long long)i * 256 * 7168, h2B, 4608, 7168, 256);
      k_add_ln<<<576, 256, 0, stream>>>(seB, h2B, rln_g + i * 256, rln_b + i * 256,
                                        X, 4608);
    }
  }

  // ---- similarity + fusion head (batch rows padded 8 -> 16) ----
  k_mean_n<<<8, 256, 0, stream>>>(satX, sg);
  k_mean_n<<<8, 256, 0, stream>>>(uavX, ug);
  k_make_si<<<32, 256, 0, stream>>>(sg, ug, siB);
  k_expand<<<NB(16LL * 512), 256, 0, stream>>>(siB, siE, 16LL * 512);
  gemm(siE, Ws1, h1h, 16, 3584, 1024);
  k_gelu<<<NB(16LL * 1024), 256, 0, stream>>>(h1h, 16LL * 1024);
  k_expand<<<NB(16LL * 1024), 256, 0, stream>>>(h1h, E2h, 16LL * 1024);
  gemm(E2h, Ws2, h2h, 16, 7168, 64);
  k_gelu<<<NB(16LL * 64), 256, 0, stream>>>(h2h, 16LL * 64);
  k_simw3<<<8, 32, 0, stream>>>(h2h, s3_base, s3_spl, wB);
  k_make_fi<<<32, 256, 0, stream>>>(sg, ug, wB, fiB);
  k_expand<<<NB(16LL * 512), 256, 0, stream>>>(fiB, fiE, 16LL * 512);
  gemm(fiE, Wfu, fusO, 16, 3584, 256);
  k_final<<<1, 256, 0, stream>>>(fusO, fln_g, fln_b, (float*)d_out);
}
